// CrossModalAttention_49804440764467
// MI455X (gfx1250) — matrix operand order repaired and checked by reading, not compiled, not run on hardware
//
#include <hip/hip_runtime.h>
#include <hip/hip_bf16.h>

typedef __attribute__((ext_vector_type(16))) _Float16 v16h;
typedef __attribute__((ext_vector_type(8)))  _Float16 v8h;
typedef __attribute__((ext_vector_type(8)))  float    v8f;
typedef __attribute__((ext_vector_type(4)))  unsigned int v4u;
typedef __attribute__((ext_vector_type(8)))  int      v8i;
typedef __attribute__((ext_vector_type(4)))  int      v4i;

static constexpr int Bx  = 4;
static constexpr int Cc  = 256;
static constexpr int Nn  = 4096;   // H*W
static constexpr int HID = 64;
static constexpr int QT  = 64;     // query tile
static constexpr int KT  = 64;     // key tile
static constexpr int PT  = 64;     // pixel tile (projection)

// ---------- WMMA helper ----------
static __device__ __forceinline__ v8f wmma_f16(v16h a, v16h b, v8f c) {
  return __builtin_amdgcn_wmma_f32_16x16x32_f16(
      /*neg_a=*/false, a, /*neg_b=*/false, b,
      /*c_mod=*/(short)0, c, /*reuse_a=*/false, /*reuse_b=*/false);
}

// A-frag (16x32 f16): source stored row-major A[row][k], stride rs halves.
static __device__ __forceinline__ v16h a_frag_load(const _Float16* p, int rs,
                                                   int rbase, int kbase, int lane) {
  int r  = rbase + (lane & 15);
  int kk = kbase + ((lane >> 4) << 3);
  const _Float16* q = p + (size_t)r * rs + kk;
  union { v16h v; v8h h[2]; } u;
  u.h[0] = *(const v8h*)(q);
  u.h[1] = *(const v8h*)(q + 16);
  return u.v;
}

// B-frag (32x16 f16): B[k][n] supplied via transposed storage T[n][k]
// (stride rs halves). Lane: n = nbase + lane%16; k = kbase+(lane/16)*8+{0..7}, +16+{0..7}.
static __device__ __forceinline__ v16h b_frag_load(const _Float16* t, int rs,
                                                   int nbase, int kbase, int lane) {
  int n  = nbase + (lane & 15);
  int kk = kbase + ((lane >> 4) << 3);
  const _Float16* q = t + (size_t)n * rs + kk;
  union { v16h v; v8h h[2]; } u;
  u.h[0] = *(const v8h*)(q);
  u.h[1] = *(const v8h*)(q + 16);
  return u.v;
}

// ---------- Tensor Data Mover: 2-D f16 tile Global -> LDS ----------
// Builds D# group0/group1 per CDNA5 ISA (8.3/8.4): count=1, type=2 (image),
// data_size=1 (2 bytes), 2-D tile, dims/strides in elements.
// clang-23 toolchain: 6-arg builtin (g0, g1, g2, g3, g_extra, cpol).
static __device__ __forceinline__ void tdm_load_2d(unsigned lds_addr,
                                                   unsigned long long gaddr,
                                                   unsigned dim0, unsigned dim1,
                                                   unsigned tile0, unsigned tile1,
                                                   unsigned long long stride0) {
  v4u g0;
  g0[0] = 1u;                                              // count=1, user D#
  g0[1] = lds_addr;                                        // lds_addr[31:0]
  g0[2] = (unsigned)(gaddr & 0xffffffffu);                 // global_addr lo
  g0[3] = (unsigned)((gaddr >> 32) & 0x01ffffffu)          // global_addr hi
        | (2u << 30);                                      // type=2
  v8i g1;
  g1[0] = (int)(1u << 16);                                 // data_size=2B
  g1[1] = (int)((dim0 & 0xffffu) << 16);                   // tensor_dim0 lo16
  g1[2] = (int)((dim0 >> 16) | ((dim1 & 0xffffu) << 16));  // dim0 hi / dim1 lo
  g1[3] = (int)((dim1 >> 16) | (tile0 << 16));             // dim1 hi / tile_dim0
  g1[4] = (int)(tile1 & 0xffffu);                          // tile_dim1 (tile_dim2=0)
  g1[5] = (int)(unsigned)(stride0 & 0xffffffffu);          // dim0_stride lo32
  g1[6] = (int)(unsigned)((stride0 >> 32) & 0xffffu);      // dim0_stride hi16
  g1[7] = 0;
  v4i z4 = {0, 0, 0, 0};
  v8i z8 = {0, 0, 0, 0, 0, 0, 0, 0};
  __builtin_amdgcn_tensor_load_to_lds(g0, g1, z4, z4, z8, 0);
}

// =====================================================================
// Kernel 1: fused projection.
//   P = [Wq; Wk; Wv] (384x256, f32->f16) x X_b (256 x 4096)
// Writes: Q[d=i] as (n,h), K[d=1-i] as (m,h), V[d=1-i] as (c,m), all f16.
// =====================================================================
static __device__ __forceinline__ v16h w_frag_load(int orow, int kbase, int lane,
                                                   const float* Wq, const float* Wk,
                                                   const float* Wv) {
  const float* Wsrc; int r;
  if (orow < 64)       { Wsrc = Wq; r = orow; }
  else if (orow < 128) { Wsrc = Wk; r = orow - 64; }
  else                 { Wsrc = Wv; r = orow - 128; }
  int row = r + (lane & 15);
  int kk  = kbase + ((lane >> 4) << 3);
  const float* p = Wsrc + (size_t)row * Cc + kk;
  v8f a = *(const v8f*)(p);
  v8f b = *(const v8f*)(p + 16);
  union { v16h v; _Float16 e[16]; } u;
#pragma unroll
  for (int j = 0; j < 8; ++j) { u.e[j] = (_Float16)a[j]; u.e[8 + j] = (_Float16)b[j]; }
  return u.v;
}

__global__ __launch_bounds__(256) void proj_kernel(
    const float* __restrict__ x1, const float* __restrict__ x2,
    const float* __restrict__ Wq, const float* __restrict__ bq,
    const float* __restrict__ Wk, const float* __restrict__ bk,
    const float* __restrict__ Wv, const float* __restrict__ bv,
    _Float16* __restrict__ Qbuf, _Float16* __restrict__ Kbuf,
    _Float16* __restrict__ Vbuf) {
  const int ntiles = Nn / PT;                   // 64
  int bid = blockIdx.x;
  int nt  = bid % ntiles;
  int b   = (bid / ntiles) % Bx;
  int i   = bid / (ntiles * Bx);                // input index
  const float* x = (i == 0 ? x1 : x2) + (size_t)b * Cc * Nn;

  int tid = threadIdx.x, lane = tid & 31, w = tid >> 5;

  __shared__ __align__(32) _Float16 Xt[PT][Cc + 16];

  {
    int n  = tid & 63;
    int cg = tid >> 6;
#pragma unroll 4
    for (int kk = 0; kk < 64; ++kk) {
      int c = cg * 64 + kk;
      Xt[n][c] = (_Float16)x[(size_t)c * Nn + nt * PT + n];
    }
  }
  __syncthreads();

  int half = lane >> 4, col = lane & 15;
#pragma unroll
  for (int mf = 0; mf < 3; ++mf) {              // 8 waves * 48 rows = 384
    int orow = w * 48 + mf * 16;
    v8f acc[4];
#pragma unroll
    for (int f = 0; f < 4; ++f) acc[f] = (v8f){};
#pragma unroll
    for (int ks = 0; ks < Cc / 32; ++ks) {
      v16h af = w_frag_load(orow, ks * 32, lane, Wq, Wk, Wv);
#pragma unroll
      for (int f = 0; f < 4; ++f) {
        v16h bf = b_frag_load(&Xt[0][0], Cc + 16, f * 16, ks * 32, lane);
        acc[f] = wmma_f16(af, bf, acc[f]);
      }
    }
    const float* bias; int rb;
    if (orow < 64)       { bias = bq; rb = orow; }
    else if (orow < 128) { bias = bk; rb = orow - 64; }
    else                 { bias = bv; rb = orow - 128; }
#pragma unroll
    for (int f = 0; f < 4; ++f) {
      int n = nt * PT + f * 16 + col;
      if (orow < 128) {
        _Float16* dst = (orow < 64)
            ? Qbuf + (((size_t)i       * Bx + b) * Nn + n) * HID + (orow + half * 8)
            : Kbuf + (((size_t)(1 - i) * Bx + b) * Nn + n) * HID + (orow - 64 + half * 8);
        v8h pk;
#pragma unroll
        for (int v = 0; v < 8; ++v)
          pk[v] = (_Float16)(acc[f][v] + bias[rb + half * 8 + v]);
        *(v8h*)dst = pk;
      } else {
#pragma unroll
        for (int v = 0; v < 8; ++v) {
          int c = orow - 128 + half * 8 + v;
          Vbuf[(((size_t)(1 - i) * Bx + b) * Cc + c) * Nn + n] =
              (_Float16)(acc[f][v] + bias[c]);
        }
      }
    }
  }
}

// =====================================================================
// Kernel 2: flash attention + residual, TDM-staged K/V tiles.
//   K tile: double-buffered, prefetched one iteration ahead.
//   V tile: DMA overlapped with the S-phase (WMMA + softmax).
// =====================================================================
__global__ __launch_bounds__(256) void attn_kernel(
    const float* __restrict__ x1, const float* __restrict__ x2,
    const float* __restrict__ gamma,
    const _Float16* __restrict__ Qbuf, const _Float16* __restrict__ Kbuf,
    const _Float16* __restrict__ Vbuf, float* __restrict__ out) {
  const int ntiles = Nn / QT;                   // 64
  const int NKT    = Nn / KT;                   // 64
  int bid = blockIdx.x;
  int qt  = bid % ntiles;
  int b   = (bid / ntiles) % Bx;
  int d   = bid / (ntiles * Bx);

  int tid = threadIdx.x, lane = tid & 31, w = tid >> 5;
  int qrb  = (w & 3) * 16;                      // query row base in tile
  int chb  = (w >> 2) * 128;                    // channel base
  bool sWave = (w < 4);
  int half = lane >> 4, col = lane & 15;

  const _Float16* Q = Qbuf + ((size_t)(d * Bx + b)) * Nn * HID;
  const _Float16* K = Kbuf + ((size_t)(d * Bx + b)) * Nn * HID;
  const _Float16* V = Vbuf + ((size_t)(d * Bx + b)) * Cc * Nn;
  const float* xq = (d == 0 ? x1 : x2);

  __shared__ __align__(32) _Float16 K_lds[2][KT][HID];   // 16 KB, double-buffered
  __shared__ __align__(32) _Float16 V_lds[Cc][KT];       // 32 KB
  __shared__ __align__(16) _Float16 P_lds[QT][KT + 8];   //  9 KB
  __shared__ float resc[QT];
  __shared__ float lsum[QT];

  // Q A-frags for this wave's 16 query rows (K-dim = HID = 64 -> 2 frags)
  v16h qa0 = a_frag_load(Q, HID, qt * QT + qrb, 0, lane);
  v16h qa1 = a_frag_load(Q, HID, qt * QT + qrb, 32, lane);

  v8f o_acc[8];
#pragma unroll
  for (int f = 0; f < 8; ++f) o_acc[f] = (v8f){};
  float m_run[8], l_run[8];
#pragma unroll
  for (int v = 0; v < 8; ++v) { m_run[v] = -1e30f; l_run[v] = 0.f; }

  // Prologue: TDM-prefetch K tile 0 (wave 0 only; TENSORcnt is per-wave).
  if (w == 0) {
    tdm_load_2d((unsigned)(uintptr_t)&K_lds[0][0][0],
                (unsigned long long)(uintptr_t)K,
                HID, KT, HID, KT, HID);
  }

  for (int kt = 0; kt < NKT; ++kt) {
    int buf = kt & 1;
    bool more = (kt + 1) < NKT;
    __syncthreads();     // prior O-phase done: V_lds / P_lds / K_lds[buf^1] free
    if (w == 0) {
      // V tile for this iteration (overlaps with S-phase below),
      // K tile for the next iteration (double buffer).
      tdm_load_2d((unsigned)(uintptr_t)&V_lds[0][0],
                  (unsigned long long)(uintptr_t)(V + (size_t)kt * KT),
                  KT, Cc, KT, Cc, (unsigned long long)Nn);
      if (more) {
        tdm_load_2d((unsigned)(uintptr_t)&K_lds[buf ^ 1][0][0],
                    (unsigned long long)(uintptr_t)(K + (size_t)(kt + 1) * KT * HID),
                    HID, KT, HID, KT, HID);
        __builtin_amdgcn_s_wait_tensorcnt(2);   // K(kt) landed (in-order TDM)
      } else {
        __builtin_amdgcn_s_wait_tensorcnt(1);   // only V(kt) still in flight
      }
    }
    __syncthreads();                            // K_lds[buf] visible to all

    if (sWave) {
      v8f s[4];
#pragma unroll
      for (int f = 0; f < 4; ++f) {
        v8f c = (v8f){};
        v16h kb0 = b_frag_load(&K_lds[buf][0][0], HID, f * 16, 0, lane);
        v16h kb1 = b_frag_load(&K_lds[buf][0][0], HID, f * 16, 32, lane);
        c = wmma_f16(qa0, kb0, c);
        c = wmma_f16(qa1, kb1, c);
        s[f] = c;
      }
      float tm[8];
#pragma unroll
      for (int v = 0; v < 8; ++v)
        tm[v] = fmaxf(fmaxf(s[0][v], s[1][v]), fmaxf(s[2][v], s[3][v]));
#pragma unroll
      for (int off = 1; off < 16; off <<= 1)
#pragma unroll
        for (int v = 0; v < 8; ++v)
          tm[v] = fmaxf(tm[v], __shfl_xor(tm[v], off, 16));
      float rs[8], tsum[8];
#pragma unroll
      for (int v = 0; v < 8; ++v) {
        float mnew = fmaxf(m_run[v], tm[v]);
        rs[v] = __expf(m_run[v] - mnew);
        m_run[v] = mnew;
        tsum[v] = 0.f;
      }
#pragma unroll
      for (int f = 0; f < 4; ++f)
#pragma unroll
        for (int v = 0; v < 8; ++v) {
          float p = __expf(s[f][v] - m_run[v]);
          s[f][v] = p;
          tsum[v] += p;
        }
#pragma unroll
      for (int off = 1; off < 16; off <<= 1)
#pragma unroll
        for (int v = 0; v < 8; ++v) tsum[v] += __shfl_xor(tsum[v], off, 16);
#pragma unroll
      for (int v = 0; v < 8; ++v) l_run[v] = l_run[v] * rs[v] + tsum[v];
#pragma unroll
      for (int f = 0; f < 4; ++f)
#pragma unroll
        for (int v = 0; v < 8; ++v)
          P_lds[qrb + half * 8 + v][f * 16 + col] = (_Float16)s[f][v];
      if (col == 0) {
#pragma unroll
        for (int v = 0; v < 8; ++v) resc[qrb + half * 8 + v] = rs[v];
      }
    }
    if (w == 0) {
      if (more) __builtin_amdgcn_s_wait_tensorcnt(1);   // V(kt) landed
      else      __builtin_amdgcn_s_wait_tensorcnt(0);
    }
    __syncthreads();                            // P, resc, V_lds ready

    float rs2[8];
#pragma unroll
    for (int v = 0; v < 8; ++v) rs2[v] = resc[qrb + half * 8 + v];
#pragma unroll
    for (int f = 0; f < 8; ++f)
#pragma unroll
      for (int v = 0; v < 8; ++v) o_acc[f][v] *= rs2[v];

    v16h pa0 = a_frag_load(&P_lds[0][0], KT + 8, qrb, 0, lane);
    v16h pa1 = a_frag_load(&P_lds[0][0], KT + 8, qrb, 32, lane);
#pragma unroll
    for (int f = 0; f < 8; ++f) {
      v16h vb0 = b_frag_load(&V_lds[0][0], KT, chb + f * 16, 0, lane);
      v16h vb1 = b_frag_load(&V_lds[0][0], KT, chb + f * 16, 32, lane);
      o_acc[f] = wmma_f16(pa0, vb0, o_acc[f]);
      o_acc[f] = wmma_f16(pa1, vb1, o_acc[f]);
    }
  }

  __syncthreads();
  if (sWave && col == 0) {
#pragma unroll
    for (int v = 0; v < 8; ++v) lsum[qrb + half * 8 + v] = l_run[v];
  }
  __syncthreads();
  float g = gamma[0];
#pragma unroll
  for (int v = 0; v < 8; ++v) {
    int qrow = qrb + half * 8 + v;
    int q    = qt * QT + qrow;
    float linv = 1.f / lsum[qrow];
#pragma unroll
    for (int f = 0; f < 8; ++f) {
      int c = chb + f * 16 + col;
      size_t oidx = ((size_t)d * Bx + b) * Cc * Nn + (size_t)c * Nn + q;
      size_t xidx = ((size_t)b * Cc + c) * Nn + q;
      out[oidx] = xq[xidx] + g * (o_acc[f][v] * linv);
    }
  }
}

extern "C" void kernel_launch(void* const* d_in, const int* in_sizes, int n_in,
                              void* d_out, int out_size, void* d_ws, size_t ws_size,
                              hipStream_t stream) {
  const float* x1 = (const float*)d_in[0];
  const float* x2 = (const float*)d_in[1];
  const float* Wq = (const float*)d_in[2];
  const float* bq = (const float*)d_in[3];
  const float* Wk = (const float*)d_in[4];
  const float* bk = (const float*)d_in[5];
  const float* Wv = (const float*)d_in[6];
  const float* bv = (const float*)d_in[7];
  const float* gamma = (const float*)d_in[8];
  float* out = (float*)d_out;

  _Float16* Qbuf = (_Float16*)d_ws;
  _Float16* Kbuf = Qbuf + (size_t)2 * Bx * Nn * HID;
  _Float16* Vbuf = Kbuf + (size_t)2 * Bx * Nn * HID;

  dim3 blk(256);
  dim3 grid_proj(2 * Bx * (Nn / PT));   // 512
  dim3 grid_attn(2 * Bx * (Nn / QT));   // 512

  proj_kernel<<<grid_proj, blk, 0, stream>>>(x1, x2, Wq, bq, Wk, bk, Wv, bv,
                                             Qbuf, Kbuf, Vbuf);
  attn_kernel<<<grid_attn, blk, 0, stream>>>(x1, x2, gamma, Qbuf, Kbuf, Vbuf, out);
}
